// SharedBasisConvBank_34832184770701
// MI455X (gfx1250) — matrix operand order repaired and checked by reading, not compiled, not run on hardware
//
#include <hip/hip_runtime.h>
#include <hip/hip_bf16.h>
#include <cstdint>

// Problem dims (compile-time)
#define B_ 256
#define D_ 1024
#define F_ 4096
#define K_ 8
#define S_ 8

// GEMM tiling
#define MT 128   // rows (b) per workgroup
#define NT 64    // cols (f or d) per workgroup
#define KT 32    // K-slice per WMMA (16x16x32 f16)

typedef __attribute__((ext_vector_type(16))) _Float16 v16h;
typedef __attribute__((ext_vector_type(8)))  float    v8f;
typedef int v4i_b128 __attribute__((vector_size(16)));   // pointee type of async b128 builtin

// ---------------------------------------------------------------------------
// Async global->LDS path (gfx1250 GLOBAL_LOAD_ASYNC_TO_LDS_B128, ASYNCcnt).
// Builtin signature (from compiler diagnostic):
//   (int4 AS1* src_global, int4 AS3* dst_lds, imm int offset, imm int cpol)
// Guarded by __has_builtin so the file compiles on toolchains without it.
// ---------------------------------------------------------------------------
#if defined(__HIP_DEVICE_COMPILE__) && __has_builtin(__builtin_amdgcn_global_load_async_to_lds_b128)
#define USE_ASYNC_LDS 1
#else
#define USE_ASYNC_LDS 0
#endif

__device__ __forceinline__ void copy16_to_lds(const _Float16* g, _Float16* l) {
#if USE_ASYNC_LDS
    __builtin_amdgcn_global_load_async_to_lds_b128(
        (__attribute__((address_space(1))) v4i_b128*)(g),
        (__attribute__((address_space(3))) v4i_b128*)(l),
        /*imm offset=*/0, /*cpol=*/0);
#else
    *reinterpret_cast<uint4*>(l) = *reinterpret_cast<const uint4*>(g);
#endif
}

#if USE_ASYNC_LDS
#  if __has_builtin(__builtin_amdgcn_s_wait_asynccnt)
#    define WAIT_ASYNC(n) __builtin_amdgcn_s_wait_asynccnt(n)
#  else
#    define WAIT_ASYNC(n) asm volatile("s_wait_asynccnt %0" :: "i"(n) : "memory")
#  endif
#else
#  define WAIT_ASYNC(n) ((void)0)
#endif

// ---------------------------------------------------------------------------
// WMMA helper: D = A(16x32 f16) * B(32x16 f16) + C(16x16 f32)
// ---------------------------------------------------------------------------
__device__ __forceinline__ v8f wmma_f16(v16h a, v16h b, v8f c) {
    return __builtin_amdgcn_wmma_f32_16x16x32_f16(
        /*neg_a=*/false, a, /*neg_b=*/false, b,
        /*c_mod=*/(short)0, c, /*reuse_a=*/false, /*reuse_b=*/false);
}

// ---------------------------------------------------------------------------
// A-fragment (16x32, 16-bit), LDS tile row-major [16][KT] (K contiguous).
// Layout: lanes 0-15 -> M=lane, K=0..15 contiguous;
//         lanes 16-31 -> M=lane-16, K=16..31 (matches B fragment).
// Dword view per lane: dwords (hi*8 .. hi*8+7).
// ---------------------------------------------------------------------------
__device__ __forceinline__ v16h load_a_frag(const _Float16* sm, int lane) {
    const int m  = lane & 15;
    const int hi = lane >> 4;
    const uint4* row = reinterpret_cast<const uint4*>(sm + m * KT); // 4 uint4/row
    union { uint4 q[2]; v16h h; } a;
    a.q[0] = row[hi * 2];       // dwords hi*8 .. hi*8+3
    a.q[1] = row[hi * 2 + 1];   // dwords hi*8+4 .. hi*8+7
    return a.h;
}

// ---------------------------------------------------------------------------
// B-fragment (32x16, 16-bit), staged in LDS transposed: [16 n][KT k] so each
// lane's K-pairs are contiguous. Layout used here: lanes 0-15 hold K=0..15
// (VGPRi: K=2i,2i+1), lanes 16-31 hold K=16..31.
// Dword view per lane: dwords (hi*8 .. hi*8+7).
// ---------------------------------------------------------------------------
__device__ __forceinline__ v16h load_b_frag(const _Float16* sm, int lane) {
    const int n  = lane & 15;
    const int hi = lane >> 4;
    const uint4* row = reinterpret_cast<const uint4*>(sm + n * KT);
    union { uint4 q[2]; v16h h; } b;
    b.q[0] = row[hi * 2];
    b.q[1] = row[hi * 2 + 1];
    return b.h;
}

// ---------------------------------------------------------------------------
// Kernel 0a: f32 -> f16 convert (x)
// ---------------------------------------------------------------------------
__global__ __launch_bounds__(256) void cvt_f16_kernel(const float* __restrict__ in,
                                                      _Float16* __restrict__ out, int n) {
    int i = blockIdx.x * 256 + threadIdx.x;
    if (i < n) out[i] = (_Float16)in[i];
}

// ---------------------------------------------------------------------------
// Kernel 0b: per-s effective weights.
// out[s*total + i] = f16( sum_k coeff[s*K+k] * w[k*total + i] )
// ---------------------------------------------------------------------------
__global__ __launch_bounds__(256) void mix_weights_kernel(const float* __restrict__ w,
                                                          const float* __restrict__ coeff,
                                                          _Float16* __restrict__ out,
                                                          int total) {
    __shared__ float sc[S_ * K_];
    if (threadIdx.x < S_ * K_) sc[threadIdx.x] = coeff[threadIdx.x];
    __syncthreads();
    int i = blockIdx.x * 256 + threadIdx.x;
    if (i >= total) return;
    float v[K_];
#pragma unroll
    for (int k = 0; k < K_; ++k) v[k] = w[(size_t)k * total + i];
#pragma unroll
    for (int s = 0; s < S_; ++s) {
        float a = 0.f;
#pragma unroll
        for (int k = 0; k < K_; ++k) a += sc[s * K_ + k] * v[k];
        out[(size_t)s * total + i] = (_Float16)a;
    }
}

// ---------------------------------------------------------------------------
// Kernel 1: fused gate/up GEMM + SiLU per s, double-buffered async LDS.
//   gate[b,f] = sum_d x16[b,d] * Wg[s,f,d];  up likewise
//   hid[s,b,f] = f16( silu(gate) * up )
// Grid: (F_/NT, B_/MT, S_). Block: 256 threads = 8 waves, wave grid 4(m)x2(n),
// each wave -> 32x32 outputs = 2x2 fragments (x2 matrices).
// ---------------------------------------------------------------------------
__global__ __launch_bounds__(256) void gemm_gateup_kernel(const _Float16* __restrict__ xg,
                                                          const _Float16* __restrict__ Wg,
                                                          const _Float16* __restrict__ Wu,
                                                          _Float16* __restrict__ hid) {
    __shared__ __align__(16) _Float16 sX[2][MT * KT];
    __shared__ __align__(16) _Float16 sG[2][NT * KT];
    __shared__ __align__(16) _Float16 sU[2][NT * KT];

    const int fBase = blockIdx.x * NT;
    const int bBase = blockIdx.y * MT;
    const int s     = blockIdx.z;

    const int tid  = threadIdx.x;
    const int lane = tid & 31;
    const int wave = tid >> 5;
    const int wm   = wave >> 1;   // 0..3  -> m offset wm*32
    const int wn   = wave & 1;    // 0..1  -> n offset wn*32

    const v8f vzero = {0.f, 0.f, 0.f, 0.f, 0.f, 0.f, 0.f, 0.f};
    v8f accG[2][2], accU[2][2];
#pragma unroll
    for (int i = 0; i < 2; ++i)
#pragma unroll
        for (int j = 0; j < 2; ++j) { accG[i][j] = vzero; accU[i][j] = vzero; }

    const _Float16* wgBase = Wg + (size_t)s * F_ * D_;
    const _Float16* wuBase = Wu + (size_t)s * F_ * D_;

    // Per-thread tile-load coordinates (fixed across stages)
    const int xm0 = tid >> 2;                 // sX row for rep 0
    const int xm1 = (tid + 256) >> 2;         // sX row for rep 1
    const int xc  = (tid & 3) << 3;           // f16 column offset within KT
    const int wn_row = tid >> 2;              // weight row
    const int wc     = (tid & 3) << 3;

    // Issue one k-stage of async (or sync) tile loads into buffer `buf`.
    auto issue = [&](int k0, int buf) {
        copy16_to_lds(xg + (size_t)(bBase + xm0) * D_ + k0 + xc,
                      &sX[buf][(size_t)(tid)       * 8]);
        copy16_to_lds(xg + (size_t)(bBase + xm1) * D_ + k0 + xc,
                      &sX[buf][(size_t)(tid + 256) * 8]);
        copy16_to_lds(wgBase + (size_t)(fBase + wn_row) * D_ + k0 + wc,
                      &sG[buf][(size_t)tid * 8]);
        copy16_to_lds(wuBase + (size_t)(fBase + wn_row) * D_ + k0 + wc,
                      &sU[buf][(size_t)tid * 8]);
        // L2 prefetch of the stage after next (global_prefetch_b8)
        if (k0 + KT < D_) {
            __builtin_prefetch(wgBase + (size_t)(fBase + wn_row) * D_ + k0 + KT + wc, 0, 1);
            __builtin_prefetch(wuBase + (size_t)(fBase + wn_row) * D_ + k0 + KT + wc, 0, 1);
        }
    };

    issue(0, 0);
    const int steps = D_ / KT;
    for (int ks = 0; ks < steps; ++ks) {
        const int cur = ks & 1;
        if (ks + 1 < steps) {
            issue((ks + 1) * KT, cur ^ 1);
            WAIT_ASYNC(4);      // 4 async ops per stage per wave -> prev stage done
        } else {
            WAIT_ASYNC(0);
        }
        __syncthreads();        // all waves' deposits for `cur` visible

        v16h a0 = load_a_frag(&sX[cur][(wm * 32 + 0)  * KT], lane);
        v16h a1 = load_a_frag(&sX[cur][(wm * 32 + 16) * KT], lane);
        v16h g0 = load_b_frag(&sG[cur][(wn * 32 + 0)  * KT], lane);
        v16h g1 = load_b_frag(&sG[cur][(wn * 32 + 16) * KT], lane);
        v16h u0 = load_b_frag(&sU[cur][(wn * 32 + 0)  * KT], lane);
        v16h u1 = load_b_frag(&sU[cur][(wn * 32 + 16) * KT], lane);

        accG[0][0] = wmma_f16(a0, g0, accG[0][0]);
        accG[0][1] = wmma_f16(a0, g1, accG[0][1]);
        accG[1][0] = wmma_f16(a1, g0, accG[1][0]);
        accG[1][1] = wmma_f16(a1, g1, accG[1][1]);
        accU[0][0] = wmma_f16(a0, u0, accU[0][0]);
        accU[0][1] = wmma_f16(a0, u1, accU[0][1]);
        accU[1][0] = wmma_f16(a1, u0, accU[1][0]);
        accU[1][1] = wmma_f16(a1, u1, accU[1][1]);

        __syncthreads();        // done reading `cur` before it is refilled
    }

    // ---- epilogue: hidden = silu(gate) * up, store f16 ----
    const int hi   = lane >> 4;
    const int colL = lane & 15;
#pragma unroll
    for (int i = 0; i < 2; ++i) {
#pragma unroll
        for (int j = 0; j < 2; ++j) {
            int f = fBase + wn * 32 + j * 16 + colL;
#pragma unroll
            for (int r = 0; r < 8; ++r) {
                int b = bBase + wm * 32 + i * 16 + hi * 8 + r;
                float g = accG[i][j][r];
                float u = accU[i][j][r];
                float h = (g / (1.0f + __expf(-g))) * u;
                hid[((size_t)s * B_ + b) * F_ + f] = (_Float16)h;
            }
        }
    }
}

// ---------------------------------------------------------------------------
// Kernel 2: down GEMM per s, double-buffered async LDS.
//   out[b, s, d] = sum_f hid[s,b,f] * Wd[s,d,f]
// Grid: (D_/NT, B_/MT, S_); same wave tiling, single weight matrix.
// ---------------------------------------------------------------------------
__global__ __launch_bounds__(256) void gemm_down_kernel(const _Float16* __restrict__ hid,
                                                        const _Float16* __restrict__ Wd,
                                                        float* __restrict__ out) {
    __shared__ __align__(16) _Float16 sH[2][MT * KT];
    __shared__ __align__(16) _Float16 sW[2][NT * KT];

    const int dBase = blockIdx.x * NT;
    const int bBase = blockIdx.y * MT;
    const int s     = blockIdx.z;

    const int tid  = threadIdx.x;
    const int lane = tid & 31;
    const int wave = tid >> 5;
    const int wm   = wave >> 1;
    const int wn   = wave & 1;

    const v8f vzero = {0.f, 0.f, 0.f, 0.f, 0.f, 0.f, 0.f, 0.f};
    v8f acc[2][2];
#pragma unroll
    for (int i = 0; i < 2; ++i)
#pragma unroll
        for (int j = 0; j < 2; ++j) acc[i][j] = vzero;

    const _Float16* hBase = hid + (size_t)s * B_ * F_;
    const _Float16* wBase = Wd + (size_t)s * D_ * F_;

    const int xm0 = tid >> 2;
    const int xm1 = (tid + 256) >> 2;
    const int xc  = (tid & 3) << 3;
    const int wn_row = tid >> 2;
    const int wc     = (tid & 3) << 3;

    auto issue = [&](int k0, int buf) {
        copy16_to_lds(hBase + (size_t)(bBase + xm0) * F_ + k0 + xc,
                      &sH[buf][(size_t)(tid)       * 8]);
        copy16_to_lds(hBase + (size_t)(bBase + xm1) * F_ + k0 + xc,
                      &sH[buf][(size_t)(tid + 256) * 8]);
        copy16_to_lds(wBase + (size_t)(dBase + wn_row) * F_ + k0 + wc,
                      &sW[buf][(size_t)tid * 8]);
        if (k0 + KT < F_) {
            __builtin_prefetch(wBase + (size_t)(dBase + wn_row) * F_ + k0 + KT + wc, 0, 1);
        }
    };

    issue(0, 0);
    const int steps = F_ / KT;
    for (int ks = 0; ks < steps; ++ks) {
        const int cur = ks & 1;
        if (ks + 1 < steps) {
            issue((ks + 1) * KT, cur ^ 1);
            WAIT_ASYNC(3);      // 3 async ops per stage per wave
        } else {
            WAIT_ASYNC(0);
        }
        __syncthreads();

        v16h a0 = load_a_frag(&sH[cur][(wm * 32 + 0)  * KT], lane);
        v16h a1 = load_a_frag(&sH[cur][(wm * 32 + 16) * KT], lane);
        v16h b0 = load_b_frag(&sW[cur][(wn * 32 + 0)  * KT], lane);
        v16h b1 = load_b_frag(&sW[cur][(wn * 32 + 16) * KT], lane);

        acc[0][0] = wmma_f16(a0, b0, acc[0][0]);
        acc[0][1] = wmma_f16(a0, b1, acc[0][1]);
        acc[1][0] = wmma_f16(a1, b0, acc[1][0]);
        acc[1][1] = wmma_f16(a1, b1, acc[1][1]);

        __syncthreads();
    }

    const int hi   = lane >> 4;
    const int colL = lane & 15;
#pragma unroll
    for (int i = 0; i < 2; ++i) {
#pragma unroll
        for (int j = 0; j < 2; ++j) {
            int d = dBase + wn * 32 + j * 16 + colL;
#pragma unroll
            for (int r = 0; r < 8; ++r) {
                int b = bBase + wm * 32 + i * 16 + hi * 8 + r;
                out[((size_t)b * S_ + s) * D_ + d] = acc[i][j][r];
            }
        }
    }
}

// ---------------------------------------------------------------------------
// Launch
// ---------------------------------------------------------------------------
extern "C" void kernel_launch(void* const* d_in, const int* in_sizes, int n_in,
                              void* d_out, int out_size, void* d_ws, size_t ws_size,
                              hipStream_t stream) {
    const float* x  = (const float*)d_in[0];
    const float* gw = (const float*)d_in[1];
    const float* uw = (const float*)d_in[2];
    const float* dw = (const float*)d_in[3];
    const float* gc = (const float*)d_in[4];
    const float* uc = (const float*)d_in[5];
    const float* dc = (const float*)d_in[6];
    float* out = (float*)d_out;

    // Workspace layout (all 16B aligned)
    const size_t X_BYTES = (size_t)B_ * D_ * sizeof(_Float16);          // 0.5 MB
    const size_t W_BYTES = (size_t)S_ * F_ * D_ * sizeof(_Float16);     // 64 MB each
    char* ws = (char*)d_ws;
    _Float16* xg  = (_Float16*)(ws);
    _Float16* Wg  = (_Float16*)(ws + X_BYTES);
    _Float16* Wu  = (_Float16*)(ws + X_BYTES + W_BYTES);
    _Float16* Wd  = (_Float16*)(ws + X_BYTES + 2 * W_BYTES);
    _Float16* hid = (_Float16*)(ws + X_BYTES + 3 * W_BYTES);            // 16 MB

    // 0) convert x, build per-s effective f16 weights
    {
        int n = B_ * D_;
        cvt_f16_kernel<<<(n + 255) / 256, 256, 0, stream>>>(x, xg, n);
        int total = F_ * D_;
        int blocks = (total + 255) / 256;
        mix_weights_kernel<<<blocks, 256, 0, stream>>>(gw, gc, Wg, total);
        mix_weights_kernel<<<blocks, 256, 0, stream>>>(uw, uc, Wu, total);
        mix_weights_kernel<<<blocks, 256, 0, stream>>>(dw, dc, Wd, total);
    }

    // 1) fused gate/up GEMM + SiLU -> hidden (f16)
    {
        dim3 grid(F_ / NT, B_ / MT, S_);
        gemm_gateup_kernel<<<grid, 256, 0, stream>>>(xg, Wg, Wu, hid);
    }

    // 2) down GEMM -> out (f32)
    {
        dim3 grid(D_ / NT, B_ / MT, S_);
        gemm_down_kernel<<<grid, 256, 0, stream>>>(hid, Wd, out);
    }
}
